// Attention_25812753449357
// MI455X (gfx1250) — matrix operand order repaired and checked by reading, not compiled, not run on hardware
//
#include <hip/hip_runtime.h>
#include <hip/hip_bf16.h>

// Causal multi-head attention, flash-attention style, bf16 WMMA / f32 softmax.
// SQ=2048, B=4, NP=16, HN=128. q/k/v layout: [s, b, n, h] ; out: [s, b, n*h].
// Per block: 128 query rows, 4 waves x 32 rows. KTILE=32 keys per iteration,
// next K/V tile prefetched into registers while computing on current LDS tile.

#define SQL   2048
#define BB    4
#define NPH   16
#define HN    128
#define QTILE 128
#define KTILE 32
#define NTHREADS 128   // 4 waves
#define MSUB  2        // two 16-row M sub-tiles per wave

// LDS row strides (bf16 elements); multiples of 8 for 16B-aligned b128 loads,
// non-multiples of 64 to spread rows across banks.
#define QSTR  136
#define KSTR  136
#define VTSTR 40
#define PSTR  40

typedef __attribute__((ext_vector_type(16))) __bf16 v16bf;
typedef __attribute__((ext_vector_type(8)))  float  v8f;

__global__ __launch_bounds__(NTHREADS, 1)
void fa_causal_kernel(const float* __restrict__ Q, const float* __restrict__ K,
                      const float* __restrict__ V, float* __restrict__ O)
{
    __shared__ __bf16 Qs[QTILE * QSTR];         // 128 x 128 row major   (34816 B)
    __shared__ __bf16 Ks[KTILE * KSTR];         // 32 x 128 row major    ( 8704 B)
    __shared__ __bf16 Vst[HN * VTSTR];          // 128 x 32 transposed   (10240 B)
    __shared__ __bf16 Ps[4 * 32 * PSTR];        // per-wave 32x32 probs  (10240 B)
                                                // total 64000 B

    const int tid  = threadIdx.x;
    const int wave = tid >> 5;
    const int lane = tid & 31;
    const int half = lane >> 4;
    const int ln16 = lane & 15;

    const int q0 = blockIdx.x * QTILE;
    const int bh = blockIdx.y;                          // b*NP + n
    const size_t rowstride = (size_t)BB * NPH * HN;     // 8192 floats / seq step
    const float* Qg = Q + (size_t)bh * HN;
    const float* Kg = K + (size_t)bh * HN;
    const float* Vg = V + (size_t)bh * HN;
    float*       Og = O + (size_t)bh * HN;

    // ---- stage Q tile once (f32 -> bf16) ----
    for (int i = tid; i < QTILE * (HN / 4); i += NTHREADS) {
        const int row = i >> 5;             // 32 float4 per row
        const int c4  = i & 31;
        const float4 v = *(const float4*)(Qg + (size_t)(q0 + row) * rowstride + c4 * 4);
        __bf16* d = &Qs[row * QSTR + c4 * 4];
        d[0] = (__bf16)v.x; d[1] = (__bf16)v.y; d[2] = (__bf16)v.z; d[3] = (__bf16)v.w;
    }

    // ---- per-wave state: 32 query rows = 2 x 16-row sub-tiles ----
    const int wrow0 = q0 + wave * 32;
    float m_i[MSUB][8], l_i[MSUB][8];
    v8f   o[MSUB][8];
    v8f   zero = {};
#pragma unroll
    for (int ms = 0; ms < MSUB; ++ms) {
#pragma unroll
        for (int r = 0; r < 8; ++r) { m_i[ms][r] = -1e30f; l_i[ms][r] = 0.0f; }
#pragma unroll
        for (int t = 0; t < 8; ++t) o[ms][t] = zero;
    }

    const float scale = 0.08838834764831843f;   // 1/sqrt(128)
    const int ktEnd = q0 + QTILE;               // causal bound

    // K/V tile register prefetch: 1024 float4 / 128 threads = 8 each
    float4 kreg[8], vreg[8];
    auto load_tile = [&](int kt0) {
#pragma unroll
        for (int j = 0; j < 8; ++j) {
            const int i   = tid + j * NTHREADS;
            const int row = i >> 5;
            const int c4  = i & 31;
            kreg[j] = *(const float4*)(Kg + (size_t)(kt0 + row) * rowstride + c4 * 4);
            vreg[j] = *(const float4*)(Vg + (size_t)(kt0 + row) * rowstride + c4 * 4);
        }
    };
    auto store_tile = [&]() {
#pragma unroll
        for (int j = 0; j < 8; ++j) {
            const int i   = tid + j * NTHREADS;
            const int row = i >> 5;
            const int c4  = i & 31;
            __bf16* d = &Ks[row * KSTR + c4 * 4];
            d[0] = (__bf16)kreg[j].x; d[1] = (__bf16)kreg[j].y;
            d[2] = (__bf16)kreg[j].z; d[3] = (__bf16)kreg[j].w;
            Vst[(c4 * 4 + 0) * VTSTR + row] = (__bf16)vreg[j].x;
            Vst[(c4 * 4 + 1) * VTSTR + row] = (__bf16)vreg[j].y;
            Vst[(c4 * 4 + 2) * VTSTR + row] = (__bf16)vreg[j].z;
            Vst[(c4 * 4 + 3) * VTSTR + row] = (__bf16)vreg[j].w;
        }
    };

    load_tile(0);

    for (int kt0 = 0; kt0 < ktEnd; kt0 += KTILE) {
        store_tile();          // regs (tile kt0) -> LDS
        __syncthreads();
        if (kt0 + KTILE < ktEnd) load_tile(kt0 + KTILE);   // prefetch next tile

        // whole-wave causal skip (kt0 > wrow0 means all 32 rows above diagonal)
        if (kt0 <= wrow0) {
            // ---- S = Q * K^T : MSUB x 2 16x16 f32 tiles ----
            v8f s[MSUB][2];
#pragma unroll
            for (int ms = 0; ms < MSUB; ++ms) { s[ms][0] = zero; s[ms][1] = zero; }
#pragma unroll
            for (int kk = 0; kk < HN; kk += 32) {
                v16bf b0, b1;
                // B fragment: B[k][n] = K[key=16g+n][h=kk+k], k = 8*half + e (e<8), 16 + 8*half + e-8 (e>=8)
                const __bf16* kr0 = &Ks[(ln16) * KSTR + kk + 8 * half];
                const __bf16* kr1 = &Ks[(16 + ln16) * KSTR + kk + 8 * half];
#pragma unroll
                for (int e = 0; e < 8; ++e) { b0[e] = kr0[e]; b0[e + 8] = kr0[e + 16]; b1[e] = kr1[e]; b1[e + 8] = kr1[e + 16]; }
#pragma unroll
                for (int ms = 0; ms < MSUB; ++ms) {
                    v16bf a;
                    const __bf16* qr = &Qs[(wave * 32 + ms * 16 + ln16) * QSTR + kk + 8 * half];
#pragma unroll
                    for (int e = 0; e < 8; ++e) { a[e] = qr[e]; a[e + 8] = qr[e + 16]; }
                    s[ms][0] = __builtin_amdgcn_wmma_f32_16x16x32_bf16(false, a, false, b0, (short)0, s[ms][0], false, false);
                    s[ms][1] = __builtin_amdgcn_wmma_f32_16x16x32_bf16(false, a, false, b1, (short)0, s[ms][1], false, false);
                }
            }

            // ---- scale + mask + online softmax per sub-tile ----
#pragma unroll
            for (int ms = 0; ms < MSUB; ++ms) {
#pragma unroll
                for (int r = 0; r < 8; ++r) {
                    const int row = wrow0 + ms * 16 + r + 8 * half;
                    float x0 = s[ms][0][r] * scale;
                    float x1 = s[ms][1][r] * scale;
                    if (kt0 + ln16 > row)      x0 = -10000.0f;
                    if (kt0 + 16 + ln16 > row) x1 = -10000.0f;

                    float rm = fmaxf(x0, x1);
                    rm = fmaxf(rm, __shfl_xor(rm, 1, 32));
                    rm = fmaxf(rm, __shfl_xor(rm, 2, 32));
                    rm = fmaxf(rm, __shfl_xor(rm, 4, 32));
                    rm = fmaxf(rm, __shfl_xor(rm, 8, 32));
                    const float mnew = fmaxf(m_i[ms][r], rm);
                    const float corr = __expf(m_i[ms][r] - mnew);
                    m_i[ms][r] = mnew;
                    const float p0 = __expf(x0 - mnew);
                    const float p1 = __expf(x1 - mnew);
                    float rs = p0 + p1;
                    rs += __shfl_xor(rs, 1, 32);
                    rs += __shfl_xor(rs, 2, 32);
                    rs += __shfl_xor(rs, 4, 32);
                    rs += __shfl_xor(rs, 8, 32);
                    l_i[ms][r] = l_i[ms][r] * corr + rs;
#pragma unroll
                    for (int t = 0; t < 8; ++t) o[ms][t][r] *= corr;
                    // probs -> per-wave LDS scratch in C-tile layout
                    __bf16* pr = &Ps[(wave * 32 + ms * 16 + r + 8 * half) * PSTR];
                    pr[ln16]      = (__bf16)p0;
                    pr[16 + ln16] = (__bf16)p1;
                }
            }
            asm volatile("s_wait_dscnt 0" ::: "memory");   // wave-local P visibility

            // ---- O += P * V ----
            v16bf pa[MSUB];
#pragma unroll
            for (int ms = 0; ms < MSUB; ++ms) {
                const __bf16* pr = &Ps[(wave * 32 + ms * 16 + ln16) * PSTR + 8 * half];
#pragma unroll
                for (int e = 0; e < 8; ++e) { pa[ms][e] = pr[e]; pa[ms][e + 8] = pr[e + 16]; }
            }
#pragma unroll
            for (int t = 0; t < 8; ++t) {
                v16bf bv;
                // B[k][n] = V[key=k][h=16t+n] = Vst[(16t+n)*VTSTR + k], k = 8*half + e (e<8), 16 + 8*half + e-8 (e>=8)
                const __bf16* vr = &Vst[(16 * t + ln16) * VTSTR + 8 * half];
#pragma unroll
                for (int e = 0; e < 8; ++e) { bv[e] = vr[e]; bv[e + 8] = vr[e + 16]; }
#pragma unroll
                for (int ms = 0; ms < MSUB; ++ms)
                    o[ms][t] = __builtin_amdgcn_wmma_f32_16x16x32_bf16(false, pa[ms], false, bv, (short)0, o[ms][t], false, false);
            }
        }
        __syncthreads();       // all reads done before next store_tile
    }

    // ---- normalize and store ----
#pragma unroll
    for (int ms = 0; ms < MSUB; ++ms) {
#pragma unroll
        for (int r = 0; r < 8; ++r) {
            const float inv = 1.0f / l_i[ms][r];
            const int row = wrow0 + ms * 16 + r + 8 * half;
            float* orow = Og + (size_t)row * rowstride;
#pragma unroll
            for (int t = 0; t < 8; ++t) orow[16 * t + ln16] = o[ms][t][r] * inv;
        }
    }
}

extern "C" void kernel_launch(void* const* d_in, const int* in_sizes, int n_in,
                              void* d_out, int out_size, void* d_ws, size_t ws_size,
                              hipStream_t stream) {
    (void)in_sizes; (void)n_in; (void)d_ws; (void)ws_size; (void)out_size;
    const float* Q = (const float*)d_in[0];
    const float* K = (const float*)d_in[1];
    const float* V = (const float*)d_in[2];
    // d_in[3] = attention_mask (causal) -- reproduced analytically in-kernel
    float* O = (float*)d_out;
    dim3 grid(SQL / QTILE, BB * NPH);
    dim3 block(NTHREADS);
    fa_causal_kernel<<<grid, block, 0, stream>>>(Q, K, V, O);
}
